// SimpleH2Model_36532991820315
// MI455X (gfx1250) — matrix operand order repaired and checked by reading, not compiled, not run on hardware
//
#include <hip/hip_runtime.h>

// ---------------------------------------------------------------------------
// SimpleH2Model forward for MI455X (gfx1250), wave32 + WMMA bf16.
//   B=32, S=1024, D_IN=D_H=D_OUT=512, heads=2 (dh=256), K_NEI=3.
// All dense GEMMs run through v_wmma_f32_16x16x32_bf16 (f32 accumulate).
// Tiles staged to LDS as bf16 via vectorized float4 loads (global_load_b128 +
// ds_store_b64 fast path); next K-chunk prefetched with global_prefetch_b8.
// KNN graph stage is done sparsely (top-4 + symmetric scatter) instead of the
// reference's dense [B,S,S] adjacency GEMM.
// ---------------------------------------------------------------------------

#define B_   32
#define S_   1024
#define DH_  512
#define HD_  256          // per-head dim
#define LN_EPS 1e-5f

typedef __attribute__((ext_vector_type(16))) __bf16 v16bf;
typedef __attribute__((ext_vector_type(8)))  float  v8f;

union BF16Frag { v16bf v; unsigned short u[16]; };

__device__ __forceinline__ unsigned short f2bf(float f) {
  unsigned int u = __float_as_uint(f);
  u += 0x7FFFu + ((u >> 16) & 1u);          // round-to-nearest-even
  return (unsigned short)(u >> 16);
}

__device__ __forceinline__ unsigned int pack2bf(float a, float b) {
  return (unsigned int)f2bf(a) | ((unsigned int)f2bf(b) << 16);
}

// ---------------------------------------------------------------------------
// Core WMMA GEMM:  C[M,N] = act( A[M,K] * W^T + bias (+ residual) )
//   A: row-major, lda.   W: if w_kn==0 -> [N,K] row-major (K contiguous),
//                        if w_kn==1 -> [K,N] row-major (transposed on LDS fill)
// Block = 128 threads (4 waves). Block tile 64Mx64N, K-chunk 64 (2 WMMA steps).
// Wave w owns rows [16w,16w+16) and all 4 N-subtiles (reuses its A fragment).
// ---------------------------------------------------------------------------
__device__ __forceinline__ void gemm_core(
    const float* __restrict__ A, long lda,
    const float* __restrict__ W, long ldw, int w_kn,
    float* __restrict__ C, long ldc,
    int M, int N, int K,
    const float* __restrict__ bias,
    const float* __restrict__ resid, long ldr,
    int relu)
{
  constexpr int BM = 64, BN = 64, KC = 64, LDSS = KC + 8; // pad vs bank conflicts
  __shared__ alignas(16) unsigned short sA[BM * LDSS];
  __shared__ alignas(16) unsigned short sB[BN * LDSS];

  const int tid  = threadIdx.x;     // 0..127
  const int wave = tid >> 5;        // 0..3
  const int lane = tid & 31;
  const int l16  = lane & 15;
  const int half = lane >> 4;

  const int m0 = blockIdx.y * BM;
  const int n0 = blockIdx.x * BN;
  const bool fullM = (m0 + BM <= M);
  const bool fullN = (n0 + BN <= N);

  v8f acc[4];
#pragma unroll
  for (int t = 0; t < 4; t++)
#pragma unroll
    for (int e = 0; e < 8; e++) acc[t][e] = 0.0f;

  for (int k0 = 0; k0 < K; k0 += KC) {
    const bool fullK = (k0 + KC <= K);
    __syncthreads();

    // ---- stage A tile (fp32 -> bf16), [BM][KC] ----
    if (fullM && fullK) {
      // vector path: 64 rows x 16 float4 = 1024 float4, 8 per thread
#pragma unroll
      for (int e = tid; e < BM * (KC / 4); e += 128) {
        int r = e >> 4, c = (e & 15) * 4;
        const float4 f = *(const float4*)(A + (long)(m0 + r) * lda + k0 + c);
        uint2 pk; pk.x = pack2bf(f.x, f.y); pk.y = pack2bf(f.z, f.w);
        *(uint2*)&sA[r * LDSS + c] = pk;
      }
    } else {
      for (int e = tid; e < BM * KC; e += 128) {
        int r = e >> 6, c = e & 63;
        int gm = m0 + r, gk = k0 + c;
        float v = (gm < M && gk < K) ? A[(long)gm * lda + gk] : 0.0f;
        sA[r * LDSS + c] = f2bf(v);
      }
    }

    // ---- stage W tile as [N][K] (transpose on the fly if W is [K,N]) ----
    if (fullN && fullK && !w_kn) {
#pragma unroll
      for (int e = tid; e < BN * (KC / 4); e += 128) {
        int r = e >> 4, c = (e & 15) * 4;
        const float4 f = *(const float4*)(W + (long)(n0 + r) * ldw + k0 + c);
        uint2 pk; pk.x = pack2bf(f.x, f.y); pk.y = pack2bf(f.z, f.w);
        *(uint2*)&sB[r * LDSS + c] = pk;
      }
    } else if (fullN && fullK && w_kn) {
      // W is [K,N]: 64 k-rows x 16 float4 across n, scatter-transpose to LDS
#pragma unroll
      for (int e = tid; e < KC * (BN / 4); e += 128) {
        int kk = e >> 4, c = (e & 15) * 4;
        const float4 f = *(const float4*)(W + (long)(k0 + kk) * ldw + n0 + c);
        sB[(c + 0) * LDSS + kk] = f2bf(f.x);
        sB[(c + 1) * LDSS + kk] = f2bf(f.y);
        sB[(c + 2) * LDSS + kk] = f2bf(f.z);
        sB[(c + 3) * LDSS + kk] = f2bf(f.w);
      }
    } else {
      for (int e = tid; e < BN * KC; e += 128) {
        int r = e >> 6, c = e & 63;
        int gn = n0 + r, gk = k0 + c;
        float v = 0.0f;
        if (gn < N && gk < K)
          v = w_kn ? W[(long)gk * ldw + gn] : W[(long)gn * ldw + gk];
        sB[r * LDSS + c] = f2bf(v);
      }
    }

    // prefetch next K-chunk while this one is consumed (global_prefetch_b8)
    if (k0 + KC < K) {
      int pr = tid & 63;
      if (m0 + pr < M)
        __builtin_prefetch(A + (long)(m0 + pr) * lda + k0 + KC, 0, 1);
      if (!w_kn && n0 + pr < N)
        __builtin_prefetch(W + (long)(n0 + pr) * ldw + k0 + KC, 0, 1);
      if (w_kn && k0 + KC + pr < K)
        __builtin_prefetch(W + (long)(k0 + KC + pr) * ldw + n0, 0, 1);
    }
    __syncthreads();

    // ---- two 16x16x32 WMMA steps over the 64-wide K chunk ----
#pragma unroll
    for (int sub = 0; sub < KC; sub += 32) {
      // A fragment: 16x32 bf16, lane = M (mod 16), ISA layout
      BF16Frag af;
      const int am = wave * 16 + l16;
#pragma unroll
      for (int i = 0; i < 8; i++) {
        int kb = sub + (i >> 2) * 16 + half * 8 + (i & 3) * 2;
        af.u[2 * i]     = sA[am * LDSS + kb];
        af.u[2 * i + 1] = sA[am * LDSS + kb + 1];
      }
#pragma unroll
      for (int t = 0; t < 4; t++) {
        // B fragment: 32x16 bf16, lane = N (mod 16), ISA layout (same K order as A)
        BF16Frag bfr;
        const int bn = t * 16 + l16;
#pragma unroll
        for (int i = 0; i < 8; i++) {
          int kb = sub + (i >> 2) * 16 + half * 8 + (i & 3) * 2;
          bfr.u[2 * i]     = sB[bn * LDSS + kb];
          bfr.u[2 * i + 1] = sB[bn * LDSS + kb + 1];
        }
        acc[t] = __builtin_amdgcn_wmma_f32_16x16x32_bf16(
            false, af.v, false, bfr.v, (short)0, acc[t], false, false);
      }
    }
  }

  // store: VGPR r holds M = r + half*8 ; lane%16 = N
#pragma unroll
  for (int t = 0; t < 4; t++) {
    int gn = n0 + t * 16 + l16;
    if (gn >= N) continue;
    float bv = bias ? bias[gn] : 0.0f;
#pragma unroll
    for (int r = 0; r < 8; r++) {
      int gm = m0 + wave * 16 + half * 8 + r;
      if (gm >= M) continue;
      float v = acc[t][r] + bv;
      if (resid) v += resid[(long)gm * ldr + gn];
      if (relu) v = v > 0.0f ? v : 0.0f;
      C[(long)gm * ldc + gn] = v;
    }
  }
}

// ---------------- GEMM kernel wrappers ----------------
__global__ void k_gemm(const float* A, long lda, const float* W, long ldw,
                       int w_kn, float* C, long ldc, int M, int N, int K,
                       const float* bias, const float* resid, long ldr, int relu) {
  gemm_core(A, lda, W, ldw, w_kn, C, ldc, M, N, K, bias, resid, ldr, relu);
}

// scores[z] = q_p · k_p^T for pair p = p0 + z  (b = p/2, h = p%2)
__global__ void k_attn_scores(const float* qkv, float* scores, int p0) {
  int p = p0 + blockIdx.z, b = p >> 1, h = p & 1;
  const float* q = qkv + (long)b * S_ * 1536 + h * HD_;
  const float* k = qkv + (long)b * S_ * 1536 + 512 + h * HD_;
  float* c = scores + (long)blockIdx.z * S_ * S_;
  gemm_core(q, 1536, k, 1536, 0, c, S_, S_, S_, HD_, nullptr, nullptr, 0, 0);
}

// ctx[b, :, h*256 : ] = attn · v   (v is [K=S, N=dh] -> w_kn=1)
__global__ void k_attn_ctx(const float* qkv, const float* attn, float* ctx, int p0) {
  int p = p0 + blockIdx.z, b = p >> 1, h = p & 1;
  const float* v = qkv + (long)b * S_ * 1536 + 1024 + h * HD_;
  const float* a = attn + (long)blockIdx.z * S_ * S_;
  float* c = ctx + (long)b * S_ * DH_ + h * HD_;
  gemm_core(a, S_, v, 1536, 1, c, DH_, S_, HD_, S_, nullptr, nullptr, 0, 0);
}

// sim[z] = nrm_b · nrm_b^T  for batch b = b0 + z
__global__ void k_sim(const float* nrm, float* sim, int b0) {
  int bb = b0 + blockIdx.z;
  const float* a = nrm + (long)bb * S_ * DH_;
  float* c = sim + (long)blockIdx.z * S_ * S_;
  gemm_core(a, DH_, a, DH_, 0, c, S_, S_, S_, DH_, nullptr, nullptr, 0, 0);
}

// ---------------- elementwise / reduction kernels ----------------
__global__ void k_softmax(float* sc, float scale) {
  float* r = sc + ((long)blockIdx.z * S_ + blockIdx.x) * S_;
  __shared__ float red[256];
  int tid = threadIdx.x;
  float v[4], mx = -3.4e38f;
#pragma unroll
  for (int i = 0; i < 4; i++) { v[i] = r[tid + i * 256] * scale; mx = fmaxf(mx, v[i]); }
  red[tid] = mx; __syncthreads();
  for (int s = 128; s > 0; s >>= 1) { if (tid < s) red[tid] = fmaxf(red[tid], red[tid + s]); __syncthreads(); }
  mx = red[0]; __syncthreads();
  float sum = 0.0f;
#pragma unroll
  for (int i = 0; i < 4; i++) { v[i] = __expf(v[i] - mx); sum += v[i]; }
  red[tid] = sum; __syncthreads();
  for (int s = 128; s > 0; s >>= 1) { if (tid < s) red[tid] += red[tid + s]; __syncthreads(); }
  float inv = 1.0f / red[0];
#pragma unroll
  for (int i = 0; i < 4; i++) r[tid + i * 256] = v[i] * inv;
}

__global__ void k_add_ln(const float* a, const float* b, const float* g,
                         const float* bt, float* out) {
  long row = blockIdx.x;
  const float* pa = a + row * DH_;
  const float* pb = b + row * DH_;
  float* po = out + row * DH_;
  int tid = threadIdx.x;
  float v0 = pa[tid] + pb[tid], v1 = pa[tid + 256] + pb[tid + 256];
  __shared__ float red[256];
  red[tid] = v0 + v1; __syncthreads();
  for (int s = 128; s > 0; s >>= 1) { if (tid < s) red[tid] += red[tid + s]; __syncthreads(); }
  float mean = red[0] * (1.0f / DH_); __syncthreads();
  float c0 = v0 - mean, c1 = v1 - mean;
  red[tid] = c0 * c0 + c1 * c1; __syncthreads();
  for (int s = 128; s > 0; s >>= 1) { if (tid < s) red[tid] += red[tid + s]; __syncthreads(); }
  float rs = rsqrtf(red[0] * (1.0f / DH_) + LN_EPS);
  po[tid]       = c0 * rs * g[tid]       + bt[tid];
  po[tid + 256] = c1 * rs * g[tid + 256] + bt[tid + 256];
}

__global__ void k_rownorm(const float* x, float* out) {
  long row = blockIdx.x;
  const float* px = x + row * DH_;
  float* po = out + row * DH_;
  int tid = threadIdx.x;
  float v0 = px[tid], v1 = px[tid + 256];
  __shared__ float red[256];
  red[tid] = v0 * v0 + v1 * v1; __syncthreads();
  for (int s = 128; s > 0; s >>= 1) { if (tid < s) red[tid] += red[tid + s]; __syncthreads(); }
  float inv = 1.0f / fmaxf(sqrtf(red[0]), 1e-12f);
  po[tid] = v0 * inv; po[tid + 256] = v1 * inv;
}

__device__ __forceinline__ void top4_insert(float v, int j, float* tv, int* ti) {
  if (v <= tv[3]) return;
  int p = 3;
  while (p > 0 && v > tv[p - 1]) { tv[p] = tv[p - 1]; ti[p] = ti[p - 1]; p--; }
  tv[p] = v; ti[p] = j;
}

__global__ void k_top4(const float* sim, int* idx) {
  int z = blockIdx.z, i = blockIdx.x, lane = threadIdx.x;
  const float* row = sim + ((long)z * S_ + i) * S_;
  float tv[4] = {-3.4e38f, -3.4e38f, -3.4e38f, -3.4e38f};
  int ti[4] = {-1, -1, -1, -1};
  for (int j = lane; j < S_; j += 32) top4_insert(row[j], j, tv, ti);
  __shared__ float sv[128];
  __shared__ int   si[128];
#pragma unroll
  for (int t = 0; t < 4; t++) { sv[lane * 4 + t] = tv[t]; si[lane * 4 + t] = ti[t]; }
  __syncthreads();
  if (lane == 0) {
    float fv[4] = {-3.4e38f, -3.4e38f, -3.4e38f, -3.4e38f};
    int fi[4] = {-1, -1, -1, -1};
    for (int e = 0; e < 128; e++) top4_insert(sv[e], si[e], fv, fi);
    int* o = idx + ((long)z * S_ + i) * 4;
#pragma unroll
    for (int t = 0; t < 4; t++) o[t] = fi[t];
  }
}

// Symmetric-KNN sparse aggregation: nsum[i] += sim(i,j)*x1[j] over edges.
__global__ void k_scatter(const float* sim, const int* idx, const float* x1,
                          float* nsum, float* cnt, int b0) {
  int z = blockIdx.z, i = blockIdx.x, tid = threadIdx.x;
  int bb = b0 + z;
  const float* simrow = sim + ((long)z * S_ + i) * S_;
  const int* myidx = idx + ((long)z * S_ + i) * 4;
#pragma unroll
  for (int t = 0; t < 4; t++) {
    int j = myidx[t];
    if (j < 0 || j == i) continue;
    float w = simrow[j];
    // forward edge i <- j (always in mask)
    for (int d = tid; d < DH_; d += 128)
      atomicAdd(&nsum[((long)bb * S_ + i) * DH_ + d],
                w * x1[((long)bb * S_ + j) * DH_ + d]);
    if (tid == 0) atomicAdd(&cnt[(long)bb * S_ + i], w);
    // reverse edge j <- i, only if j's own top-k misses i (dedupe)
    const int* jidx = idx + ((long)z * S_ + j) * 4;
    bool has = false;
#pragma unroll
    for (int tt = 0; tt < 4; tt++) has = has || (jidx[tt] == i);
    if (!has) {
      for (int d = tid; d < DH_; d += 128)
        atomicAdd(&nsum[((long)bb * S_ + j) * DH_ + d],
                  w * x1[((long)bb * S_ + i) * DH_ + d]);
      if (tid == 0) atomicAdd(&cnt[(long)bb * S_ + j], w);
    }
  }
}

__global__ void k_combined(const float* x1, const float* nsum, const float* cnt,
                           float* comb, long total) {
  long gid = (long)blockIdx.x * blockDim.x + threadIdx.x;
  if (gid >= total) return;
  long m = gid >> 10;          // /1024
  int d = (int)(gid & 1023);
  comb[gid] = (d < DH_) ? x1[m * DH_ + d]
                        : nsum[m * DH_ + (d - DH_)] / fmaxf(cnt[m], 1.0f);
}

__global__ void k_pool(const float* x2, float* pooled) {
  int gid = blockIdx.x * blockDim.x + threadIdx.x;   // 0..B_*DH_-1
  int b = gid >> 9, d = gid & 511;
  const float* base = x2 + ((long)b * S_) * DH_ + d;
  float s = 0.0f;
  for (int t = 0; t < S_; t++) s += base[(long)t * DH_];
  pooled[gid] = s * (1.0f / S_);
}

// ---------------------------------------------------------------------------
extern "C" void kernel_launch(void* const* d_in, const int* in_sizes, int n_in,
                              void* d_out, int out_size, void* d_ws, size_t ws_size,
                              hipStream_t stream) {
  (void)in_sizes; (void)n_in; (void)out_size; (void)ws_size;
  const float* x      = (const float*)d_in[0];
  const float* enc_w  = (const float*)d_in[1];
  const float* enc_b  = (const float*)d_in[2];
  const float* in_w   = (const float*)d_in[3];
  const float* in_b   = (const float*)d_in[4];
  const float* out_w  = (const float*)d_in[5];
  const float* out_b  = (const float*)d_in[6];
  const float* ln1_g  = (const float*)d_in[7];
  const float* ln1_b  = (const float*)d_in[8];
  const float* ffn_w1 = (const float*)d_in[9];
  const float* ffn_b1 = (const float*)d_in[10];
  const float* ffn_w2 = (const float*)d_in[11];
  const float* ffn_b2 = (const float*)d_in[12];
  const float* gc_w   = (const float*)d_in[13];
  const float* gc_b   = (const float*)d_in[14];
  const float* ln2_g  = (const float*)d_in[15];
  const float* ln2_b  = (const float*)d_in[16];
  const float* dec_w  = (const float*)d_in[17];
  const float* dec_b  = (const float*)d_in[18];
  float* out = (float*)d_out;

  const long BS = (long)B_ * S_;                 // 32768 rows
  float* ws = (float*)d_ws;
  // workspace layout (floats); big regions recycled across phases
  const long L_H = BS * DH_, L_QKV = BS * 1536, L_MID = BS * 1024;
  const long L_SC = 8L * S_ * S_;
  float* h       = ws;                            // [BS,512]   (later: nrm)
  float* qkv     = h + L_H;                       // [BS,1536]  (later: x2)
  float* ctx     = qkv + L_QKV;                   // [BS,512]   (later: nsum)
  float* attnout = ctx + L_H;                     // [BS,512]   (later: gc_out)
  float* x1      = attnout + L_H;                 // [BS,512]
  float* mid     = x1 + L_H;                      // [BS,1024]  (later: combined)
  float* scores  = mid + L_MID;                   // 8 x [S,S]
  float* simbuf  = scores + L_SC;                 // 8 x [S,S]
  float* cnt     = simbuf + L_SC;                 // [BS]
  float* pooled  = cnt + BS;                      // [B,512]
  int*   idxbuf  = (int*)(pooled + (long)B_ * DH_); // 8 x [S,4]
  float* nrm = h;  float* nsum = ctx;  float* gcout = attnout;  float* x2 = qkv;

  const dim3 blk(128);

  // 1. encoder: h = relu(x @ enc_w^T + enc_b)
  k_gemm<<<dim3(8, 512), blk, 0, stream>>>(x, 512, enc_w, 512, 0, h, 512,
                                           32768, 512, 512, enc_b, nullptr, 0, 1);
  // 2. qkv = h @ in_w^T + in_b
  k_gemm<<<dim3(24, 512), blk, 0, stream>>>(h, 512, in_w, 512, 0, qkv, 1536,
                                            32768, 1536, 512, in_b, nullptr, 0, 0);
  // 3. attention, 8 (b,h) pairs at a time through the 32MB scores buffer
  for (int c = 0; c < 8; c++) {
    int p0 = c * 8;
    k_attn_scores<<<dim3(16, 16, 8), blk, 0, stream>>>(qkv, scores, p0);
    k_softmax<<<dim3(1024, 1, 8), 256, 0, stream>>>(scores, 0.0625f); // 1/sqrt(256)
    k_attn_ctx<<<dim3(4, 16, 8), blk, 0, stream>>>(qkv, scores, ctx, p0);
  }
  // 4. out projection
  k_gemm<<<dim3(8, 512), blk, 0, stream>>>(ctx, 512, out_w, 512, 0, attnout, 512,
                                           32768, 512, 512, out_b, nullptr, 0, 0);
  // 5. x1 = LN(h + attn_out)
  k_add_ln<<<32768, 256, 0, stream>>>(h, attnout, ln1_g, ln1_b, x1);
  // 6. FFN (residual folded into second GEMM)
  k_gemm<<<dim3(16, 512), blk, 0, stream>>>(x1, 512, ffn_w1, 512, 0, mid, 1024,
                                            32768, 1024, 512, ffn_b1, nullptr, 0, 1);
  k_gemm<<<dim3(8, 512), blk, 0, stream>>>(mid, 1024, ffn_w2, 1024, 0, x1, 512,
                                           32768, 512, 1024, ffn_b2, x1, 512, 0);
  // 7. cosine-sim KNN graph (sparse): nrm, sim per batch chunk, top4, scatter
  k_rownorm<<<32768, 256, 0, stream>>>(x1, nrm);
  hipMemsetAsync(nsum, 0, (size_t)L_H * sizeof(float), stream);
  hipMemsetAsync(cnt, 0, (size_t)BS * sizeof(float), stream);
  for (int c = 0; c < 4; c++) {
    int b0 = c * 8;
    k_sim<<<dim3(16, 16, 8), blk, 0, stream>>>(nrm, simbuf, b0);
    k_top4<<<dim3(1024, 1, 8), 32, 0, stream>>>(simbuf, idxbuf);
    k_scatter<<<dim3(1024, 1, 8), 128, 0, stream>>>(simbuf, idxbuf, x1, nsum, cnt, b0);
  }
  // 8. combined = [x1, nsum/max(cnt,1)] ; gc projection; LN2
  k_combined<<<131072, 256, 0, stream>>>(x1, nsum, cnt, mid, BS * 1024L);
  k_gemm<<<dim3(8, 512), blk, 0, stream>>>(mid, 1024, gc_w, 1024, 0, gcout, 512,
                                           32768, 512, 1024, gc_b, nullptr, 0, 0);
  k_add_ln<<<32768, 256, 0, stream>>>(x1, gcout, ln2_g, ln2_b, x2);
  // 9. mean-pool over S, then decoder
  k_pool<<<64, 256, 0, stream>>>(x2, pooled);
  k_gemm<<<dim3(8, 1), blk, 0, stream>>>(pooled, 512, dec_w, 512, 0, out, 512,
                                         32, 512, 512, dec_b, nullptr, 0, 0);
}
